// RadarConv_14087492731376
// MI455X (gfx1250) — matrix operand order repaired and checked by reading, not compiled, not run on hardware
//
#include <hip/hip_runtime.h>
#include <hip/hip_bf16.h>

// ---------------------------------------------------------------------------
// Deformable-conv pipeline for MI455X (gfx1250), wave32 + WMMA implicit GEMM.
//   B=8, Cin=16, Cout=64, H=W=128, 3x3, stride=1, pad=1
//   1) h      = conv3x3(x, W1)+b1                 (GEMM M=131072 N=64 K=144)
//   2) off/msk= conv3x3(h, [Wo;Wm]) fused         (GEMM M=131072 N=27->32 K=576)
//   3) out    = deform_conv(h, off, msk, W2)+b2   (GEMM M=131072 N=64 K=576)
// All GEMMs use v_wmma_f32_16x16x32_f16 (f16 in / f32 accum).
// Gathers are branchless (clamped address + cndmask'd value / zeroed weight);
// K -> (channel, tap) decomposition via LDS LUTs (wave-uniform broadcast).
// ---------------------------------------------------------------------------

typedef __attribute__((ext_vector_type(16))) _Float16 v16h;
typedef __attribute__((ext_vector_type(8)))  float    v8f;

#define Hdim 128
#define Wdim 128
#define HW   16384   // 128*128
#define C1   16
#define C2   64

// A-fragment K index (16-bit 16x32 A, ISA 7.12.2): row M = lane&15;
// lanes 0-15: halves 0..7 -> K 0..7, halves 8..15 -> K 16..23; lanes 16-31 +8.
__device__ __forceinline__ int kofA(int h, int lane) {
    return (h < 8 ? h : h + 8) + ((lane >> 4) & 1) * 8;
}
// Contiguous K index (unused; B is packed with kofA): lanes 0-15 K=h, lanes 16-31 K=h+16.
__device__ __forceinline__ int kofB(int h, int lane) {
    return h + ((lane >> 4) & 1) * 16;
}

// ---------------------------------------------------------------------------
// Weight packing into the exact B-fragment layout: one contiguous 32B load
// per lane per fragment.  dst[((chunk*NT+nt)*32+lane)*16+h] = f16(W[N][K]).
// mode 0: W1 (Kmax=144, NT=4)  mode 1: [Wo;Wm] (Kmax=576, NT=2)  mode 2: W2.
// ---------------------------------------------------------------------------
__global__ void __launch_bounds__(256)
pack_weights(const float* __restrict__ w0, const float* __restrict__ w1,
             _Float16* __restrict__ dst, int chunks, int nt, int kmax,
             int cin, int mode) {
    int idx = blockIdx.x * blockDim.x + threadIdx.x;
    int total = chunks * nt * 32 * 16;
    if (idx >= total) return;
    int h     = idx & 15;
    int lane  = (idx >> 4) & 31;
    int rest  = idx >> 9;
    int ntIdx = rest % nt;
    int chunk = rest / nt;
    int K = chunk * 32 + kofA(h, lane);
    int N = ntIdx * 16 + (lane & 15);
    float v = 0.0f;
    if (K < kmax) {
        int ci  = K / 9;
        int tap = K - ci * 9;
        if (mode == 1) {
            if (N < 18)       v = w0[(N * cin + ci) * 9 + tap];
            else if (N < 27)  v = w1[((N - 18) * cin + ci) * 9 + tap];
        } else {
            v = w0[(N * cin + ci) * 9 + tap];   // OIHW, N < 64 by construction
        }
    }
    dst[idx] = (_Float16)v;
}

// Branchless im2col A fragment using LDS LUT: lut[K].x = (ci<<14)+(ky<<7)+kx,
// lut[K].y = (ky<<8)|kx  (ky=255 for padded K >= Kmax -> always invalid).
__device__ __forceinline__ v16h
a_im2col(const float* __restrict__ src, const int2* __restrict__ lut,
         int baseIdx, int y, int xcol, int ch, int lane) {
    v16h a;
    int kbase = ch * 32 + ((lane >> 4) & 1) * 8;
#pragma unroll
    for (int hh = 0; hh < 16; ++hh) {
        int K = kbase + (hh < 8 ? hh : hh + 8);
        int2 l = lut[K];                      // wave-uniform LDS broadcast
        int yy = y + (l.y >> 8) - 1;
        int xx = xcol + (l.y & 255) - 1;
        bool ok = ((unsigned)yy < (unsigned)Hdim) & ((unsigned)xx < (unsigned)Wdim);
        int idx = ok ? (baseIdx + l.x) : 0;   // clamped address, no branch
        float v = src[idx];
        a[hh] = (_Float16)(ok ? v : 0.0f);
    }
    return a;
}

template <int NT>
__device__ __forceinline__ void
wmma_step(const v16h& a, const _Float16* __restrict__ wp, int ch, int lane, v8f* acc) {
#pragma unroll
    for (int nt = 0; nt < NT; ++nt) {
        const _Float16* p = wp + (size_t)((ch * NT + nt) * 512 + lane * 16);
        v16h bv = *(const v16h*)p;
        acc[nt] = __builtin_amdgcn_wmma_f32_16x16x32_f16(
            false, a, false, bv, (short)0, acc[nt], false, false);
    }
}

// Build the K-decomposition LUT cooperatively (one small div per entry, once).
template <int KPAD, int KMAX>
__device__ __forceinline__ void build_lut(int2* lut) {
    for (int k = threadIdx.x; k < KPAD; k += 128) {
        int ci = k / 9, t = k - 9 * ci, ky = t / 3, kx = t - 3 * ky;
        int2 e;
        e.x = (ci << 14) + (ky << 7) + kx;
        e.y = (k < KMAX) ? ((ky << 8) | kx) : (255 << 8);
        lut[k] = e;
    }
}

// ---------------------------------------------------------------------------
// Stage 1: h = conv3x3(x)+b1. Block = 4 waves, one 16-pixel row tile per wave
// (W=128 % 16 == 0: tiles never cross an image row) x all 64 channels.
// ---------------------------------------------------------------------------
__global__ void __launch_bounds__(128)
conv1_wmma(const float* __restrict__ xin, const _Float16* __restrict__ wp,
           const float* __restrict__ bias, float* __restrict__ hout) {
    __shared__ int2 lut[160];
    build_lut<160, 144>(lut);
    __syncthreads();

    const int lane = threadIdx.x & 31;
    const int wv   = threadIdx.x >> 5;
    const int tb   = (blockIdx.x * 4 + wv) * 16;
    const int b    = tb >> 14;
    const int rem  = tb & (HW - 1);
    const int y    = rem >> 7;
    const int xb   = rem & 127;
    const int xcol = xb + (lane & 15);
    const int baseIdx = ((b * C1) << 14) + ((y - 1) << 7) + (xcol - 1);

    v8f acc[4];
#pragma unroll
    for (int i = 0; i < 4; ++i)
#pragma unroll
        for (int j = 0; j < 8; ++j) acc[i][j] = 0.0f;

#pragma unroll
    for (int ch = 0; ch < 5; ++ch) {          // K = 144 padded to 160
        v16h a = a_im2col(xin, lut, baseIdx, y, xcol, ch, lane);
        wmma_step<4>(a, wp, ch, lane, acc);
    }

    const int n0   = lane & 15;
    const int mAdd = (lane >> 4) * 8;
#pragma unroll
    for (int nt = 0; nt < 4; ++nt) {
        int co = nt * 16 + n0;
        float bi = bias[co];
        float* op = hout + (((size_t)b * C2 + co) << 14) + (y << 7) + xb;
#pragma unroll
        for (int r = 0; r < 8; ++r) op[r + mAdd] = acc[nt][r] + bi;
    }
}

// ---------------------------------------------------------------------------
// Stage 2: fused offset(18ch) + mask(9ch, 2*sigmoid) conv over h. N padded 32.
// ---------------------------------------------------------------------------
__global__ void __launch_bounds__(128)
convom_wmma(const float* __restrict__ hin, const _Float16* __restrict__ wp,
            const float* __restrict__ bo, const float* __restrict__ bm,
            float* __restrict__ offp, float* __restrict__ mskp) {
    __shared__ int2 lut[576];
    build_lut<576, 576>(lut);
    __syncthreads();

    const int lane = threadIdx.x & 31;
    const int wv   = threadIdx.x >> 5;
    const int tb   = (blockIdx.x * 4 + wv) * 16;
    const int b    = tb >> 14;
    const int rem  = tb & (HW - 1);
    const int y    = rem >> 7;
    const int xb   = rem & 127;
    const int xcol = xb + (lane & 15);
    const int baseIdx = ((b * C2) << 14) + ((y - 1) << 7) + (xcol - 1);

    v8f acc[2];
#pragma unroll
    for (int i = 0; i < 2; ++i)
#pragma unroll
        for (int j = 0; j < 8; ++j) acc[i][j] = 0.0f;

    for (int ch = 0; ch < 18; ++ch) {         // K = 576 exact
        v16h a = a_im2col(hin, lut, baseIdx, y, xcol, ch, lane);
        wmma_step<2>(a, wp, ch, lane, acc);
    }

    const int n0   = lane & 15;
    const int mAdd = (lane >> 4) * 8;
#pragma unroll
    for (int nt = 0; nt < 2; ++nt) {
        int co = nt * 16 + n0;
#pragma unroll
        for (int r = 0; r < 8; ++r) {
            int si = (y << 7) + xb + r + mAdd;
            float v = acc[nt][r];
            if (co < 18) {
                offp[(((size_t)b * 18 + co) << 14) + si] = v + bo[co];
            } else if (co < 27) {
                int mc = co - 18;
                float s = v + bm[mc];
                mskp[(((size_t)b * 9 + mc) << 14) + si] = 2.0f / (1.0f + __expf(-s));
            }
        }
    }
}

// ---------------------------------------------------------------------------
// Stage 3: deformable conv.  Per-(pixel,tap) descriptors hold pre-clamped
// corner indices + validity/mask-folded bilinear weights (32B, b128-aligned);
// shared by the lane pair L/L+16 that owns the same A row.  Inner loop is
// fully branchless: klut + 2 LDS b128 + 4 global loads + 4 FMA per element.
// ---------------------------------------------------------------------------
struct alignas(16) Tap4 { int i00, i01, i10, i11; float w00, w01, w10, w11; };

__global__ void __launch_bounds__(128)
deform_wmma(const float* __restrict__ hin, const float* __restrict__ offp,
            const float* __restrict__ mskp, const _Float16* __restrict__ wp,
            const float* __restrict__ bias, float* __restrict__ out) {
    __shared__ int2 klut[576];                // x = c<<14, y = tap
    __shared__ Tap4 taps[4][144];             // 4 waves x 16 pixels x 9 taps

    for (int k = threadIdx.x; k < 576; k += 128) {
        int c = k / 9;
        klut[k] = make_int2(c << 14, k - 9 * c);
    }

    const int lane = threadIdx.x & 31;
    const int wv   = threadIdx.x >> 5;
    const int tb   = (blockIdx.x * 4 + wv) * 16;
    const int b    = tb >> 14;
    const int rem  = tb & (HW - 1);
    const int y    = rem >> 7;
    const int xb   = rem & 127;

    for (int e = lane; e < 144; e += 32) {
        int p  = e / 9;
        int t  = e - 9 * p;
        int px = xb + p;
        int si = (y << 7) + px;
        float oy = offp[(((size_t)b * 18 + 2 * t)     << 14) + si];
        float ox = offp[(((size_t)b * 18 + 2 * t + 1) << 14) + si];
        float mv = mskp[(((size_t)b * 9  + t)         << 14) + si];
        float py  = (float)(y  - 1 + t / 3) + oy;
        float pxx = (float)(px - 1 + t % 3) + ox;
        float fy = floorf(py), fx = floorf(pxx);
        int y0 = (int)fy, x0 = (int)fx;
        float wy = py - fy, wx = pxx - fx;
        bool yv0 = (unsigned)y0       < (unsigned)Hdim;
        bool yv1 = (unsigned)(y0 + 1) < (unsigned)Hdim;
        bool xv0 = (unsigned)x0       < (unsigned)Wdim;
        bool xv1 = (unsigned)(x0 + 1) < (unsigned)Wdim;
        int y0c = yv0 ? y0 : 0, y1c = yv1 ? (y0 + 1) : 0;
        int x0c = xv0 ? x0 : 0, x1c = xv1 ? (x0 + 1) : 0;
        float ay = 1.0f - wy, ax = 1.0f - wx;
        Tap4 tp;
        tp.i00 = (y0c << 7) + x0c;  tp.i01 = (y0c << 7) + x1c;
        tp.i10 = (y1c << 7) + x0c;  tp.i11 = (y1c << 7) + x1c;
        tp.w00 = (yv0 && xv0) ? ay * ax * mv : 0.0f;
        tp.w01 = (yv0 && xv1) ? ay * wx * mv : 0.0f;
        tp.w10 = (yv1 && xv0) ? wy * ax * mv : 0.0f;
        tp.w11 = (yv1 && xv1) ? wy * wx * mv : 0.0f;
        taps[wv][e] = tp;
    }
    __syncthreads();

    v8f acc[4];
#pragma unroll
    for (int i = 0; i < 4; ++i)
#pragma unroll
        for (int j = 0; j < 8; ++j) acc[i][j] = 0.0f;

    const float* hbb = hin + ((size_t)(b * C2) << 14);
    const int rowBase = (lane & 15) * 9;

    for (int ch = 0; ch < 18; ++ch) {         // K = 576 exact
        int kbase = ch * 32 + ((lane >> 4) & 1) * 8;
        v16h a;
#pragma unroll
        for (int hh = 0; hh < 16; ++hh) {
            int K = kbase + (hh < 8 ? hh : hh + 8);
            int2 kl = klut[K];                // wave-uniform LDS broadcast
            Tap4 tp = taps[wv][rowBase + kl.y];
            const float* hb = hbb + kl.x;     // kl.x = c<<14
            float v = hb[tp.i00] * tp.w00 + hb[tp.i01] * tp.w01
                    + hb[tp.i10] * tp.w10 + hb[tp.i11] * tp.w11;
            a[hh] = (_Float16)v;
        }
        wmma_step<4>(a, wp, ch, lane, acc);
    }

    const int n0   = lane & 15;
    const int mAdd = (lane >> 4) * 8;
#pragma unroll
    for (int nt = 0; nt < 4; ++nt) {
        int co = nt * 16 + n0;
        float bi = bias[co];
        float* op = out + (((size_t)b * C2 + co) << 14) + (y << 7) + xb;
#pragma unroll
        for (int r = 0; r < 8; ++r) op[r + mAdd] = acc[nt][r] + bi;
    }
}

// ---------------------------------------------------------------------------
// Workspace layout (bytes, all 256-aligned):
//   h: 33554432   off: 9437184   mask: 4718592
//   wp1: 20480  wpom: 36864  wp2: 73728   total ~45.6 MB
// ---------------------------------------------------------------------------
extern "C" void kernel_launch(void* const* d_in, const int* in_sizes, int n_in,
                              void* d_out, int out_size, void* d_ws, size_t ws_size,
                              hipStream_t stream) {
    const float* x  = (const float*)d_in[0];
    const float* W1 = (const float*)d_in[1];
    const float* b1 = (const float*)d_in[2];
    const float* Wo = (const float*)d_in[3];
    const float* bo = (const float*)d_in[4];
    const float* Wm = (const float*)d_in[5];
    const float* bm = (const float*)d_in[6];
    const float* W2 = (const float*)d_in[7];
    const float* b2 = (const float*)d_in[8];
    float* out = (float*)d_out;

    char* ws = (char*)d_ws;
    float*    hbuf = (float*)(ws + 0);
    float*    offb = (float*)(ws + 33554432u);
    float*    mskb = (float*)(ws + 42991616u);
    _Float16* wp1  = (_Float16*)(ws + 47710208u);
    _Float16* wpom = (_Float16*)(ws + 47730688u);
    _Float16* wp2  = (_Float16*)(ws + 47767552u);

    pack_weights<<<40,  256, 0, stream>>>(W1, nullptr, wp1, 5, 4, 144, C1, 0);
    pack_weights<<<72,  256, 0, stream>>>(Wo, Wm, wpom, 18, 2, 576, C2, 1);
    pack_weights<<<144, 256, 0, stream>>>(W2, nullptr, wp2, 18, 4, 576, C2, 2);

    // 131072 pixels / 16 per wave / 4 waves per block = 2048 blocks (exact).
    conv1_wmma <<<2048, 128, 0, stream>>>(x, wp1, b1, hbuf);
    convom_wmma<<<2048, 128, 0, stream>>>(hbuf, wpom, bo, bm, offb, mskb);
    deform_wmma<<<2048, 128, 0, stream>>>(hbuf, offb, mskb, wp2, b2, out);
}
